// MultiHeadedSelfAttention_13005160972484
// MI455X (gfx1250) — matrix operand order repaired and checked by reading, not compiled, not run on hardware
//
#include <hip/hip_runtime.h>

// Problem constants (from reference)
constexpr int cB = 2, cS = 2048, cD = 1024, cH = 16, cHW = 64;
constexpr int cM = cB * cS;                 // 4096 rows of x

// QKV GEMM tiling
constexpr int BM = 128, BN = 64, BK = 64;
// Attention tiling
constexpr int TK = 64;                      // keys per LDS tile
// Shared LDS row stride for all 64-element bf16 tile rows: 64 + 8 = 72 ushorts
// (144B, multiple of 16B; produced by TDM pad: +4 DWORDs after every 32 DWORDs)
constexpr int LDT = 72;

typedef __bf16 bf16x16 __attribute__((ext_vector_type(16)));
typedef float  f32x8   __attribute__((ext_vector_type(8)));
typedef unsigned int v4u __attribute__((ext_vector_type(4)));
typedef int          v4i __attribute__((ext_vector_type(4)));
typedef int          v8i __attribute__((ext_vector_type(8)));

struct U128 { unsigned int d[4]; };
union Frag16 { bf16x16 v; U128 u[2]; };

__device__ __forceinline__ unsigned int f2bf(float f) {
  __bf16 h = (__bf16)f;                     // RNE convert
  return (unsigned int)__builtin_bit_cast(unsigned short, h);
}

__device__ __forceinline__ f32x8 wmma_bf16(bf16x16 a, bf16x16 b, f32x8 c) {
  // 8 args: (neg_a, A, neg_b, B, c_mod, C, reuse_a, reuse_b)
  return __builtin_amdgcn_wmma_f32_16x16x32_bf16(false, a, false, b, (short)0, c,
                                                 false, false);
}

// ---------------------------------------------------------------------------
// Tensor Data Mover support (6-arg builtin on this toolchain), manual fallback.
// ---------------------------------------------------------------------------
#if defined(__has_builtin)
#  if __has_builtin(__builtin_amdgcn_tensor_load_to_lds) && \
      __has_builtin(__builtin_amdgcn_s_wait_tensorcnt)
#    define HAVE_TDM 1
#  endif
#endif
#ifndef HAVE_TDM
#  define HAVE_TDM 0
#endif

#if HAVE_TDM
// D# group1: 2-byte elements, 2D tensor, LDS pad enabled (interval=32 DWORDs,
// amount=4 DWORDs -> 72-ushort row stride for 64-element rows).
__device__ __forceinline__ v8i tdm_g1(unsigned dim0, unsigned dim1,
                                      unsigned long long stride0,
                                      unsigned tile0, unsigned tile1) {
  v8i g;
  g[0] = (int)((1u << 16) |            // data_size = 2 bytes
               (1u << 20) |            // pad_enable
               (4u << 22) |            // pad_interval: 32 DWORDs
               (3u << 25));            // pad_amount:   4 DWORDs
  g[1] = (int)((dim0 & 0xFFFFu) << 16);                    // tensor_dim0[15:0]
  g[2] = (int)((dim0 >> 16) | ((dim1 & 0xFFFFu) << 16));   // dim0 hi | dim1 lo
  g[3] = (int)((dim1 >> 16) | (tile0 << 16));              // dim1 hi | tile_dim0
  g[4] = (int)(tile1 & 0xFFFFu);                           // tile_dim1 (tile_dim2=0)
  g[5] = (int)(unsigned)(stride0 & 0xFFFFFFFFull);         // dim0_stride[31:0]
  g[6] = (int)(unsigned)((stride0 >> 32) & 0xFFFFull);     // dim0_stride[47:32]
  g[7] = 0;
  return g;
}

__device__ __forceinline__ void tdm_load(const unsigned short* lds,
                                         const unsigned short* gmem, v8i g1) {
  const unsigned long long ga = (unsigned long long)(uintptr_t)gmem;
  v4u g0;
  g0[0] = 1u;                                              // count=1 (valid), user mode
  g0[1] = (unsigned)(uintptr_t)lds;                        // LDS byte address
  g0[2] = (unsigned)ga;                                    // global_addr[31:0]
  g0[3] = (unsigned)((ga >> 32) & 0x01FFFFFFull) | (2u << 30); // addr hi | type=2
  v4i z4 = {0, 0, 0, 0};                                   // 2D: groups 2/3 unused
  v8i z8 = {0, 0, 0, 0, 0, 0, 0, 0};
  __builtin_amdgcn_tensor_load_to_lds(g0, g1, z4, z4, z8, 0);
}
#endif

// ---------------------------------------------------------------------------
// Kernel 0: fp32 -> bf16 bulk convert (x and the three weight matrices).
// ---------------------------------------------------------------------------
__global__ __launch_bounds__(256) void to_bf16(const float* __restrict__ src,
                                               unsigned short* __restrict__ dst,
                                               int n8) {
  const int i = blockIdx.x * 256 + threadIdx.x;
  if (i >= n8) return;
  const float4* p = reinterpret_cast<const float4*>(src) + (size_t)i * 2;
  const float4 f0 = p[0], f1 = p[1];
  U128 o;
  o.d[0] = f2bf(f0.x) | (f2bf(f0.y) << 16);
  o.d[1] = f2bf(f0.z) | (f2bf(f0.w) << 16);
  o.d[2] = f2bf(f1.x) | (f2bf(f1.y) << 16);
  o.d[3] = f2bf(f1.z) | (f2bf(f1.w) << 16);
  reinterpret_cast<U128*>(dst)[i] = o;
}

// ---------------------------------------------------------------------------
// Kernel 1: fused QKV projection, bf16 WMMA, TDM-staged double-buffered tiles.
// Q/K written [B][H][S][HW]; V written transposed [B][H][HW][S]; Q pre-scaled
// by 1/sqrt(HW).
// ---------------------------------------------------------------------------
__global__ __launch_bounds__(256) void qkv_gemm(
    const unsigned short* __restrict__ xbf, const unsigned short* __restrict__ wbf,
    const float* __restrict__ bq, const float* __restrict__ bk, const float* __restrict__ bv,
    unsigned short* __restrict__ qo, unsigned short* __restrict__ ko,
    unsigned short* __restrict__ vo)
{
  const int z = blockIdx.z;
  const unsigned short* w = wbf + (size_t)z * cD * cD;
  const float* bias   = (z == 0) ? bq : (z == 1) ? bk : bv;
  unsigned short* out = (z == 0) ? qo : (z == 1) ? ko : vo;
  const float scale   = (z == 0) ? 0.125f : 1.0f;     // 1/sqrt(64) folded into Q

  __shared__ unsigned short xs[2][BM * LDT];
  __shared__ unsigned short wsm[2][BN * LDT];

  const int t = threadIdx.x;
  const int wave = t >> 5, lane = t & 31;
  const int lhalf = lane >> 4, l16 = lane & 15;
  const int m0 = blockIdx.x * BM;
  const int n0 = blockIdx.y * BN;

  const f32x8 zv = {0.f, 0.f, 0.f, 0.f, 0.f, 0.f, 0.f, 0.f};
  f32x8 acc[4] = {zv, zv, zv, zv};

  constexpr int NIT = cD / BK;                  // 16 K-steps

#if HAVE_TDM
  const v8i g1x = tdm_g1(cD, cM, cD, BK, BM);
  const v8i g1w = tdm_g1(cD, cD, cD, BK, BN);
  if (wave == 0) {
    tdm_load(&xs[0][0],  xbf + (size_t)m0 * cD, g1x);
    tdm_load(&wsm[0][0], w   + (size_t)n0 * cD, g1w);
  }
#endif

  for (int it = 0; it < NIT; ++it) {
    const int k0 = it * BK;
#if HAVE_TDM
    const int buf = it & 1;
    if (wave == 0) __builtin_amdgcn_s_wait_tensorcnt(0);
    __syncthreads();
    if (wave == 0 && it + 1 < NIT) {            // prefetch next tiles via DMA
      tdm_load(&xs[buf ^ 1][0],  xbf + (size_t)m0 * cD + k0 + BK, g1x);
      tdm_load(&wsm[buf ^ 1][0], w   + (size_t)n0 * cD + k0 + BK, g1w);
    }
#else
    const int buf = 0;
    __syncthreads();
#pragma unroll
    for (int i = 0; i < 4; ++i) {               // x tile: 128x64 bf16
      const int idx = t + i * 256, r = idx >> 3, c8 = idx & 7;
      *reinterpret_cast<U128*>(&xs[0][r * LDT + c8 * 8]) =
          *reinterpret_cast<const U128*>(&xbf[(size_t)(m0 + r) * cD + k0 + c8 * 8]);
    }
#pragma unroll
    for (int i = 0; i < 2; ++i) {               // w tile: 64x64 bf16
      const int idx = t + i * 256, r = idx >> 3, c8 = idx & 7;
      *reinterpret_cast<U128*>(&wsm[0][r * LDT + c8 * 8]) =
          *reinterpret_cast<const U128*>(&w[(size_t)(n0 + r) * cD + k0 + c8 * 8]);
    }
    __syncthreads();
#endif

    // A fragments: rows m = wave*16 + (lane&15), K split by lane half
    Frag16 a[2];
    const int mrow = wave * 16 + l16;
#pragma unroll
    for (int j = 0; j < 2; ++j) {
      a[j].u[0] = *reinterpret_cast<const U128*>(&xs[buf][mrow * LDT + j * 32 + lhalf * 8]);
      a[j].u[1] = *reinterpret_cast<const U128*>(&xs[buf][mrow * LDT + j * 32 + 16 + lhalf * 8]);
    }
    // Preload all B fragments, then a dense WMMA group
    Frag16 bfr[4][2];
#pragma unroll
    for (int nt = 0; nt < 4; ++nt) {
      const int nrow = nt * 16 + l16;
#pragma unroll
      for (int j = 0; j < 2; ++j) {
        bfr[nt][j].u[0] = *reinterpret_cast<const U128*>(&wsm[buf][nrow * LDT + j * 32 + lhalf * 8]);
        bfr[nt][j].u[1] = *reinterpret_cast<const U128*>(&wsm[buf][nrow * LDT + j * 32 + 16 + lhalf * 8]);
      }
    }
#pragma unroll
    for (int j = 0; j < 2; ++j)
#pragma unroll
      for (int nt = 0; nt < 4; ++nt)
        acc[nt] = wmma_bf16(a[j].v, bfr[nt][j].v, acc[nt]);
  }

  // Epilogue: bias, scale, scatter (V transposed to [B][H][HW][S])
#pragma unroll
  for (int nt = 0; nt < 4; ++nt) {
    const int ng = n0 + nt * 16 + l16;
    const float bb = bias[ng];
    const int hh = ng >> 6, hw = ng & 63;
#pragma unroll
    for (int r = 0; r < 8; ++r) {
      const int mg = m0 + wave * 16 + lhalf * 8 + r;
      const int bidx = mg >> 11, s = mg & (cS - 1);
      const float v = (acc[nt][r] + bb) * scale;
      const size_t oidx = (z == 2)
          ? (((size_t)(bidx * cH + hh) * cHW + hw) * cS + s)
          : (((size_t)(bidx * cH + hh) * cS + s) * cHW + hw);
      out[oidx] = (unsigned short)f2bf(v);
    }
  }
}

// ---------------------------------------------------------------------------
// Kernel 2: flash attention.  4 waves/block; each wave owns a 16-row q-tile;
// K (row-major) and V (pre-transposed) tiles DMA'd into double-buffered LDS.
// ---------------------------------------------------------------------------
__global__ __launch_bounds__(128) void attention(
    const unsigned short* __restrict__ qws, const unsigned short* __restrict__ kws,
    const unsigned short* __restrict__ vws, const int* __restrict__ mask,
    const float* __restrict__ pos, float* __restrict__ outp)
{
  __shared__ unsigned short Kt[2][TK * LDT];    // [key][hw]
  __shared__ unsigned short Vt[2][cHW * LDT];   // [hw][key]
  __shared__ unsigned short Pt[4][16 * LDT];    // per-wave probs tile

  const int t = threadIdx.x;
  const int wave = t >> 5, lane = t & 31;
  const int lhalf = lane >> 4, l16 = lane & 15;
  const int b = blockIdx.z, h = blockIdx.y;
  const int q0 = blockIdx.x * 64 + wave * 16;

  const unsigned short* Q  = qws + (size_t)(b * cH + h) * cS * cHW;   // [s][hw]
  const unsigned short* Kg = kws + (size_t)(b * cH + h) * cS * cHW;   // [s][hw]
  const unsigned short* Vg = vws + (size_t)(b * cH + h) * cHW * cS;   // [hw][s]

  // Q fragments resident in VGPRs for the whole kernel
  Frag16 aQ[2];
  {
    const int mrow = q0 + l16;
#pragma unroll
    for (int j = 0; j < 2; ++j) {
      aQ[j].u[0] = *reinterpret_cast<const U128*>(&Q[(size_t)mrow * cHW + j * 32 + lhalf * 8]);
      aQ[j].u[1] = *reinterpret_cast<const U128*>(&Q[(size_t)mrow * cHW + j * 32 + 16 + lhalf * 8]);
    }
  }

  const f32x8 zv = {0.f, 0.f, 0.f, 0.f, 0.f, 0.f, 0.f, 0.f};
  f32x8 O[4] = {zv, zv, zv, zv};
  float mrun[8], lrun[8];
#pragma unroll
  for (int r = 0; r < 8; ++r) { mrun[r] = -1e30f; lrun[r] = 0.0f; }

  unsigned short* Pw = &Pt[wave][0];
  constexpr int NIT = cS / TK;                  // 32 key tiles

#if HAVE_TDM
  const v8i g1k = tdm_g1(cHW, cS, cHW, 64, TK); // K tile: 64 hw x 64 keys
  const v8i g1v = tdm_g1(cS, cHW, cS, 64, cHW); // V tile: 64 keys x 64 hw rows
  if (wave == 0) {
    tdm_load(&Kt[0][0], Kg, g1k);
    tdm_load(&Vt[0][0], Vg, g1v);
  }
#endif

  for (int it = 0; it < NIT; ++it) {
    const int kt0 = it * TK;
#if HAVE_TDM
    const int buf = it & 1;
    if (wave == 0) __builtin_amdgcn_s_wait_tensorcnt(0);
    __syncthreads();
    if (wave == 0 && it + 1 < NIT) {            // DMA next tiles while computing
      tdm_load(&Kt[buf ^ 1][0], Kg + (size_t)(kt0 + TK) * cHW, g1k);
      tdm_load(&Vt[buf ^ 1][0], Vg + (size_t)(kt0 + TK), g1v);
    }
#else
    const int buf = 0;
    __syncthreads();
#pragma unroll
    for (int i = 0; i < 4; ++i) {               // K tile copy
      const int idx = t + i * 128, r = idx >> 3, c8 = idx & 7;
      *reinterpret_cast<U128*>(&Kt[0][r * LDT + c8 * 8]) =
          *reinterpret_cast<const U128*>(&Kg[(size_t)(kt0 + r) * cHW + c8 * 8]);
    }
#pragma unroll
    for (int i = 0; i < 4; ++i) {               // V tile copy (already transposed)
      const int idx = t + i * 128, r = idx >> 3, c8 = idx & 7;
      *reinterpret_cast<U128*>(&Vt[0][r * LDT + c8 * 8]) =
          *reinterpret_cast<const U128*>(&Vg[(size_t)r * cS + kt0 + c8 * 8]);
    }
    __syncthreads();
#endif

    // Stream next pos slab toward the caches
    if (kt0 + TK < cS)
      __builtin_prefetch(&pos[((size_t)b * cS + q0 + lhalf * 8) * cS + kt0 + TK + l16 * 4], 0, 1);

    // ---- scores S = Q @ K^T (16 x 64, f32): preload frags, dense WMMA group
    Frag16 bK[4][2];
#pragma unroll
    for (int nt = 0; nt < 4; ++nt) {
      const int krow = nt * 16 + l16;
#pragma unroll
      for (int j = 0; j < 2; ++j) {
        bK[nt][j].u[0] = *reinterpret_cast<const U128*>(&Kt[buf][krow * LDT + j * 32 + lhalf * 8]);
        bK[nt][j].u[1] = *reinterpret_cast<const U128*>(&Kt[buf][krow * LDT + j * 32 + 16 + lhalf * 8]);
      }
    }
    f32x8 sc[4];
#pragma unroll
    for (int nt = 0; nt < 4; ++nt) sc[nt] = wmma_bf16(aQ[0].v, bK[nt][0].v, zv);
#pragma unroll
    for (int nt = 0; nt < 4; ++nt) sc[nt] = wmma_bf16(aQ[1].v, bK[nt][1].v, sc[nt]);

    // ---- + pos bias, key mask ----
#pragma unroll
    for (int nt = 0; nt < 4; ++nt) {
      const int kc = kt0 + nt * 16 + l16;
      const float mv = -10000.0f * (1.0f - (float)mask[b * cS + kc]);
#pragma unroll
      for (int r = 0; r < 8; ++r) {
        const int qrow = q0 + lhalf * 8 + r;
        sc[nt][r] += pos[((size_t)b * cS + qrow) * cS + kc] + mv;
      }
    }

    // ---- online softmax (row reductions across 16-lane C-matrix groups) ----
#pragma unroll
    for (int r = 0; r < 8; ++r) {
      float rm = fmaxf(fmaxf(sc[0][r], sc[1][r]), fmaxf(sc[2][r], sc[3][r]));
#pragma unroll
      for (int mlk = 1; mlk < 16; mlk <<= 1) rm = fmaxf(rm, __shfl_xor(rm, mlk, 32));
      const float mnew = fmaxf(mrun[r], rm);
      const float corr = __expf(mrun[r] - mnew);
      mrun[r] = mnew;
      lrun[r] *= corr;
#pragma unroll
      for (int h4 = 0; h4 < 4; ++h4) O[h4][r] *= corr;
      float rs = 0.0f;
#pragma unroll
      for (int nt = 0; nt < 4; ++nt) {
        const float p = __expf(sc[nt][r] - mnew);
        sc[nt][r] = p;
        rs += p;
      }
#pragma unroll
      for (int mlk = 1; mlk < 16; mlk <<= 1) rs += __shfl_xor(rs, mlk, 32);
      lrun[r] += rs;
    }

    // ---- P (bf16) -> per-wave LDS (C-layout -> A-layout) ----
#pragma unroll
    for (int nt = 0; nt < 4; ++nt)
#pragma unroll
      for (int r = 0; r < 8; ++r)
        Pw[(lhalf * 8 + r) * LDT + nt * 16 + l16] = (unsigned short)f2bf(sc[nt][r]);

    // ---- O += P @ V ----
    Frag16 aP[2];
#pragma unroll
    for (int j = 0; j < 2; ++j) {
      aP[j].u[0] = *reinterpret_cast<const U128*>(&Pw[l16 * LDT + j * 32 + lhalf * 8]);
      aP[j].u[1] = *reinterpret_cast<const U128*>(&Pw[l16 * LDT + j * 32 + 16 + lhalf * 8]);
    }
    Frag16 bV[4][2];
#pragma unroll
    for (int h4 = 0; h4 < 4; ++h4) {
      const int vrow = h4 * 16 + l16;            // hw index (row of Vt)
#pragma unroll
      for (int j = 0; j < 2; ++j) {
        bV[h4][j].u[0] = *reinterpret_cast<const U128*>(&Vt[buf][vrow * LDT + j * 32 + lhalf * 8]);
        bV[h4][j].u[1] = *reinterpret_cast<const U128*>(&Vt[buf][vrow * LDT + j * 32 + 16 + lhalf * 8]);
      }
    }
#pragma unroll
    for (int h4 = 0; h4 < 4; ++h4) O[h4] = wmma_bf16(aP[0].v, bV[h4][0].v, O[h4]);
#pragma unroll
    for (int h4 = 0; h4 < 4; ++h4) O[h4] = wmma_bf16(aP[1].v, bV[h4][1].v, O[h4]);
  }

  // ---- normalize and store fp32 output [B][S][D] ----
#pragma unroll
  for (int h4 = 0; h4 < 4; ++h4) {
    const int hw = h4 * 16 + l16;
#pragma unroll
    for (int r = 0; r < 8; ++r) {
      const int qrow = q0 + lhalf * 8 + r;
      outp[((size_t)b * cS + qrow) * cD + h * cHW + hw] = O[h4][r] / lrun[r];
    }
  }
}

// ---------------------------------------------------------------------------
extern "C" void kernel_launch(void* const* d_in, const int* in_sizes, int n_in,
                              void* d_out, int out_size, void* d_ws, size_t ws_size,
                              hipStream_t stream) {
  const float* x    = (const float*)d_in[0];
  const int*   mask = (const int*)d_in[1];
  const float* pos  = (const float*)d_in[2];
  const float* wq   = (const float*)d_in[3];
  const float* bq   = (const float*)d_in[4];
  const float* wk   = (const float*)d_in[5];
  const float* bk   = (const float*)d_in[6];
  const float* wv   = (const float*)d_in[7];
  const float* bv   = (const float*)d_in[8];
  float* out = (float*)d_out;

  // Workspace layout (bf16 ushorts): Q | K | V | xbf | wbf(q,k,v)
  const size_t per = (size_t)cB * cS * cD;     // 4,194,304 elements
  const size_t wsz = (size_t)cD * cD;          // 1,048,576 elements
  unsigned short* qws = (unsigned short*)d_ws;
  unsigned short* kws = qws + per;
  unsigned short* vws = kws + per;
  unsigned short* xbf = vws + per;
  unsigned short* wbf = xbf + per;

  // fp32 -> bf16 conversions
  to_bf16<<<dim3((int)(per / 8 / 256)), 256, 0, stream>>>(x, xbf, (int)(per / 8));
  to_bf16<<<dim3((int)(wsz / 8 / 256)), 256, 0, stream>>>(wq, wbf, (int)(wsz / 8));
  to_bf16<<<dim3((int)(wsz / 8 / 256)), 256, 0, stream>>>(wk, wbf + wsz, (int)(wsz / 8));
  to_bf16<<<dim3((int)(wsz / 8 / 256)), 256, 0, stream>>>(wv, wbf + 2 * wsz, (int)(wsz / 8));

  qkv_gemm<<<dim3(cM / BM, cD / BN, 3), 256, 0, stream>>>(
      xbf, wbf, bq, bk, bv, qws, kws, vws);
  attention<<<dim3(cS / 64, cH, cB), 128, 0, stream>>>(
      qws, kws, vws, mask, pos, out);
}
